// SplitedTreeLSTM_25151328485755
// MI455X (gfx1250) — matrix operand order repaired and checked by reading, not compiled, not run on hardware
//
#include <hip/hip_runtime.h>

typedef __attribute__((ext_vector_type(16))) _Float16 v16h;
typedef __attribute__((ext_vector_type(8)))  _Float16 v8h;
typedef __attribute__((ext_vector_type(8)))  float    v8f;

#define NTRACE 4096
#define LPOS   64
#define ISZ    64
#define HSZ    128
#define OSZ    128

// LDS row strides (f16 units), padded for bank spread; byte stride multiple of 16
#define SX 72
#define SH 136

// workspace byte offsets (all 32B aligned)
#define WS_ACCUM 0        // 128 f32
#define WS_BIAS1 512      // 384 f32
#define WS_BIAS2 2048     // 384 f32
#define WS_W1P   4096     // 24576 f16  (2 ktiles x 24 ntiles x 512)
#define WS_W2P   53248    // 49152 f16  (4 ktiles x 24 ntiles x 512)
#define WS_WP1P  151552   // 8192 f16   (4 ktiles x 4  ntiles x 512)

__device__ __forceinline__ v8f wmma16(v16h a, v16h b, v8f c) {
  // D = A(16x32 f16) x B(32x16 f16) + C(16x16 f32)
  return __builtin_amdgcn_wmma_f32_16x16x32_f16(false, a, false, b, (short)0, c,
                                                false, false);
}

// CDNA5 hardware tanh (TRANS32). Builtin if the toolchain declares it, else raw ISA.
#if __has_builtin(__builtin_amdgcn_tanhf)
__device__ __forceinline__ float fast_tanh(float x) { return __builtin_amdgcn_tanhf(x); }
#else
__device__ __forceinline__ float fast_tanh(float x) {
  float y;
  // trailing v_nop: 1-instruction TRANS hazard gap (raw asm bypasses scheduler model)
  asm("v_tanh_f32 %0, %1\n\tv_nop" : "=v"(y) : "v"(x));
  return y;
}
#endif

__device__ __forceinline__ float sigm(float x) {
  return __builtin_fmaf(0.5f, fast_tanh(0.5f * x), 0.5f);
}

// B fragment: pre-swizzled in ws so each lane does one contiguous 32B load.
__device__ __forceinline__ v16h load_b_tile(const _Float16* __restrict__ W, int tile, int lane) {
  return *(const v16h*)(W + ((tile << 5) + lane) * 16);
}

// A fragment from LDS, row-major [M][K] f16 with stride (f16 units).
// Lane layout: M = lane%16; half = lane/16; K = half*16 + e
__device__ __forceinline__ v16h load_a_tile(const _Float16* s, int stride, int mt, int kt, int lane) {
  const int m  = mt * 16 + (lane & 15);
  const int hf = lane >> 4;
  const _Float16* p = s + m * stride + kt * 32 + hf * 16;
  v8h lo = *(const v8h*)p;
  v8h hi = *(const v8h*)(p + 8);
  return __builtin_shufflevector(lo, hi, 0,1,2,3,4,5,6,7,8,9,10,11,12,13,14,15);
}

// ---------------- prep: zero accumulator, fuse biases, pack f16 weights in B layout ---------
__global__ __launch_bounds__(256) void prep_kernel(
    const float* __restrict__ W_ih1, const float* __restrict__ b_ih1, const float* __restrict__ b_hh1,
    const float* __restrict__ W_ih2, const float* __restrict__ b_ih2, const float* __restrict__ b_hh2,
    const float* __restrict__ Wp1,
    float* __restrict__ accum, float* __restrict__ bias1, float* __restrict__ bias2,
    _Float16* __restrict__ W1p, _Float16* __restrict__ W2p, _Float16* __restrict__ Wp1p)
{
  int i = blockIdx.x * 256 + threadIdx.x;
  if (i < 128) { accum[i] = 0.0f; return; }
  i -= 128;
  if (i < 384) {  // packed gate cols: [0:128]=i, [128:256]=c_hat(orig 256:384), [256:384]=o(orig 384:512)
    int no = (i < 128) ? i : i + 128;
    bias1[i] = b_ih1[no] + b_hh1[no];
    return;
  }
  i -= 384;
  if (i < 384) {
    int no = (i < 128) ? i : i + 128;
    bias2[i] = b_ih2[no] + b_hh2[no];
    return;
  }
  i -= 384;
  if (i < 24576) {  // W1p: K=64 (2 ktiles), N=384 (24 ntiles)
    int tile = i >> 9, r = i & 511, lane = r >> 4, e = r & 15;
    int kt = tile / 24, nt = tile % 24;
    int k  = kt * 32 + (lane >> 4) * 16 + e;       // B frag: K = half*16 + e within ktile
    int np = nt * 16 + (lane & 15);
    int no = (np < 128) ? np : np + 128;
    W1p[i] = (_Float16)W_ih1[no * ISZ + k];
    return;
  }
  i -= 24576;
  if (i < 49152) {  // W2p: K=128 (4 ktiles), N=384
    int tile = i >> 9, r = i & 511, lane = r >> 4, e = r & 15;
    int kt = tile / 24, nt = tile % 24;
    int k  = kt * 32 + (lane >> 4) * 16 + e;
    int np = nt * 16 + (lane & 15);
    int no = (np < 128) ? np : np + 128;
    W2p[i] = (_Float16)W_ih2[no * HSZ + k];
    return;
  }
  i -= 49152;
  if (i < 8192) {   // Wp1p: K=128 (4 ktiles), N=64 (4 ntiles)
    int tile = i >> 9, r = i & 511, lane = r >> 4, e = r & 15;
    int kt = tile >> 2, nt = tile & 3;
    int k  = kt * 32 + (lane >> 4) * 16 + e;
    int n  = nt * 16 + (lane & 15);
    Wp1p[i] = (_Float16)Wp1[n * HSZ + k];
  }
}

// ---------------- main fused kernel: one trace per workgroup (256 thr = 8 waves) ------------
__global__ __launch_bounds__(256) void fused_trace_kernel(
    const float* __restrict__ emb, const int* __restrict__ traces, const int* __restrict__ lengths,
    const float* __restrict__ bp1, const float* __restrict__ Wp2, const float* __restrict__ bp2,
    const _Float16* __restrict__ W1p, const _Float16* __restrict__ W2p, const _Float16* __restrict__ Wp1p,
    const float* __restrict__ bias1, const float* __restrict__ bias2,
    float* __restrict__ accum)
{
  __shared__ _Float16 sX [LPOS * SX] __attribute__((aligned(16)));   // gathered emb, f16
  __shared__ _Float16 sH1[LPOS * SH] __attribute__((aligned(16)));
  __shared__ _Float16 sH2[LPOS * SH] __attribute__((aligned(16)));
  __shared__ float    sE  [LPOS];
  __shared__ float    sWgt[LPOS];
  __shared__ int      sTok[LPOS];

  const int tid  = threadIdx.x;
  const int w    = tid >> 5;
  const int lane = tid & 31;
  const int trc  = blockIdx.x;

  // ---- gather embeddings -> LDS f16 ----
  if (tid < LPOS) sTok[tid] = traces[trc * LPOS + tid];
  if (w == 7) {  // warm L2 with the biggest weight block while others gather
    for (int off = lane * 512; off < 49152; off += 32 * 512)
      __builtin_prefetch(W2p + off, 0, 1);
  }
  __syncthreads();
  for (int i = tid; i < LPOS * ISZ; i += 256) {
    int pos = i >> 6, f = i & 63;
    sX[pos * SX + f] = (_Float16)emb[sTok[pos] * ISZ + f];
  }
  __syncthreads();

  // ---- GEMM1 + gate nonlinearity: h1 = sig(o)*tanh(sig(i)*tanh(c)) ----
  {
    const int nt = w;                        // 8 waves x 8 gate-col tiles of width 16
    const int np = nt * 16 + (lane & 15);
    const float bi = bias1[np], bc = bias1[np + 128], bo = bias1[np + 256];
    for (int mt = 0; mt < 4; ++mt) {
      // fused bias rides in the WMMA C accumulator (same value for all 8 rows of a lane)
      v8f a0 = {bi, bi, bi, bi, bi, bi, bi, bi};
      v8f a1 = {bc, bc, bc, bc, bc, bc, bc, bc};
      v8f a2 = {bo, bo, bo, bo, bo, bo, bo, bo};
      #pragma unroll
      for (int kt = 0; kt < 2; ++kt) {
        v16h A = load_a_tile(sX, SX, mt, kt, lane);
        a0 = wmma16(A, load_b_tile(W1p, kt * 24 + nt,      lane), a0);
        a1 = wmma16(A, load_b_tile(W1p, kt * 24 + nt + 8,  lane), a1);
        a2 = wmma16(A, load_b_tile(W1p, kt * 24 + nt + 16, lane), a2);
      }
      const int row0 = mt * 16 + (lane >> 4) * 8;
      #pragma unroll
      for (int r = 0; r < 8; ++r) {
        float cc = sigm(a0[r]) * fast_tanh(a1[r]);
        float hh = sigm(a2[r]) * fast_tanh(cc);
        sH1[(row0 + r) * SH + np] = (_Float16)hh;
      }
    }
  }
  __syncthreads();

  // ---- GEMM2 + gate nonlinearity -> h2 ----
  {
    const int nt = w;
    const int np = nt * 16 + (lane & 15);
    const float bi = bias2[np], bc = bias2[np + 128], bo = bias2[np + 256];
    for (int mt = 0; mt < 4; ++mt) {
      v8f a0 = {bi, bi, bi, bi, bi, bi, bi, bi};
      v8f a1 = {bc, bc, bc, bc, bc, bc, bc, bc};
      v8f a2 = {bo, bo, bo, bo, bo, bo, bo, bo};
      #pragma unroll
      for (int kt = 0; kt < 4; ++kt) {
        v16h A = load_a_tile(sH1, SH, mt, kt, lane);
        a0 = wmma16(A, load_b_tile(W2p, kt * 24 + nt,      lane), a0);
        a1 = wmma16(A, load_b_tile(W2p, kt * 24 + nt + 8,  lane), a1);
        a2 = wmma16(A, load_b_tile(W2p, kt * 24 + nt + 16, lane), a2);
      }
      const int row0 = mt * 16 + (lane >> 4) * 8;
      #pragma unroll
      for (int r = 0; r < 8; ++r) {
        float cc = sigm(a0[r]) * fast_tanh(a1[r]);
        float hh = sigm(a2[r]) * fast_tanh(cc);
        sH2[(row0 + r) * SH + np] = (_Float16)hh;
      }
    }
  }
  __syncthreads();

  // ---- attention energies: relu(h2 @ Wp1.T + bp1) . Wp2 + bp2 ----
  if (tid < LPOS) sE[tid] = bp2[0];
  __syncthreads();
  for (int t = w; t < 16; t += 8) {          // 16 tiles: mt in 0..3, nt in 0..3
    const int mt = t >> 2, nt = t & 3;
    const int n  = nt * 16 + (lane & 15);
    const float bb = bp1[n], ww = Wp2[n];
    v8f acc = {bb, bb, bb, bb, bb, bb, bb, bb};
    #pragma unroll
    for (int kt = 0; kt < 4; ++kt) {
      v16h A = load_a_tile(sH2, SH, mt, kt, lane);
      acc = wmma16(A, load_b_tile(Wp1p, kt * 4 + nt, lane), acc);
    }
    const int m0 = mt * 16 + (lane >> 4) * 8;
    #pragma unroll
    for (int r = 0; r < 8; ++r) {
      float v = fmaxf(acc[r], 0.0f) * ww;
      v += __shfl_xor(v, 1);
      v += __shfl_xor(v, 2);
      v += __shfl_xor(v, 4);
      v += __shfl_xor(v, 8);
      if ((lane & 15) == 0) atomicAdd(&sE[m0 + r], v);
    }
  }
  __syncthreads();

  // ---- masked softmax over positions (wave 0; 2 positions per lane) ----
  if (w == 0) {
    const int len = lengths[trc];
    float e0 = (lane      < len) ? sE[lane]      : -1e30f;
    float e1 = (lane + 32 < len) ? sE[lane + 32] : -1e30f;
    float mx = fmaxf(e0, e1);
    for (int o = 16; o; o >>= 1) mx = fmaxf(mx, __shfl_xor(mx, o));
    float x0 = (lane      < len) ? __expf(e0 - mx) : 0.0f;
    float x1 = (lane + 32 < len) ? __expf(e1 - mx) : 0.0f;
    float s = x0 + x1;
    for (int o = 16; o; o >>= 1) s += __shfl_xor(s, o);
    float inv = 1.0f / s;
    sWgt[lane]      = x0 * inv;
    sWgt[lane + 32] = x1 * inv;
  }
  __syncthreads();

  // ---- weighted sum of h2 over positions; accumulate across traces ----
  {
    const int col = tid & 127, part = tid >> 7;
    float a = 0.0f;
    #pragma unroll 8
    for (int l = part * 32; l < part * 32 + 32; ++l)
      a += sWgt[l] * (float)sH2[l * SH + col];
    atomicAdd(&accum[col], a);
  }
}

// ---------------- epilogue: out = accum @ W_out.T + b_out ----------------
__global__ __launch_bounds__(128) void out_kernel(
    const float* __restrict__ accum, const float* __restrict__ W_out,
    const float* __restrict__ b_out, float* __restrict__ out)
{
  const int o = threadIdx.x;
  float s = b_out[o];
  #pragma unroll 8
  for (int h = 0; h < HSZ; ++h) s += accum[h] * W_out[o * HSZ + h];
  out[o] = s;
}

extern "C" void kernel_launch(void* const* d_in, const int* in_sizes, int n_in,
                              void* d_out, int out_size, void* d_ws, size_t ws_size,
                              hipStream_t stream) {
  (void)in_sizes; (void)n_in; (void)out_size; (void)ws_size;
  const float* emb    = (const float*)d_in[0];
  const float* W_ih1  = (const float*)d_in[1];
  // d_in[2] = W_hh1 (unused: h0 = 0)
  const float* b_ih1  = (const float*)d_in[3];
  const float* b_hh1  = (const float*)d_in[4];
  const float* W_ih2  = (const float*)d_in[5];
  // d_in[6] = W_hh2 (unused)
  const float* b_ih2  = (const float*)d_in[7];
  const float* b_hh2  = (const float*)d_in[8];
  const float* Wp1    = (const float*)d_in[9];
  const float* bp1    = (const float*)d_in[10];
  const float* Wp2    = (const float*)d_in[11];
  const float* bp2    = (const float*)d_in[12];
  const float* W_out  = (const float*)d_in[13];
  const float* b_out  = (const float*)d_in[14];
  const int*   traces = (const int*)d_in[15];
  const int*   lens   = (const int*)d_in[16];

  char* ws = (char*)d_ws;
  float*    accum = (float*)(ws + WS_ACCUM);
  float*    bias1 = (float*)(ws + WS_BIAS1);
  float*    bias2 = (float*)(ws + WS_BIAS2);
  _Float16* W1p   = (_Float16*)(ws + WS_W1P);
  _Float16* W2p   = (_Float16*)(ws + WS_W2P);
  _Float16* Wp1p  = (_Float16*)(ws + WS_WP1P);

  // 128 + 384 + 384 + 24576 + 49152 + 8192 = 82816 work items
  prep_kernel<<<(82816 + 255) / 256, 256, 0, stream>>>(
      W_ih1, b_ih1, b_hh1, W_ih2, b_ih2, b_hh2, Wp1,
      accum, bias1, bias2, W1p, W2p, Wp1p);

  fused_trace_kernel<<<NTRACE, 256, 0, stream>>>(
      emb, traces, lens, bp1, Wp2, bp2, W1p, W2p, Wp1p, bias1, bias2, accum);

  out_kernel<<<1, 128, 0, stream>>>(accum, W_out, b_out, (float*)d_out);
}
